// TurboQuantLinear_41772851921434
// MI455X (gfx1250) — matrix operand order repaired and checked by reading, not compiled, not run on hardware
//
#include <hip/hip_runtime.h>

// Problem sizes (fixed by the reference)
#define T_TOK   8192      // 4 * 2048 tokens
#define KDIM    4096
#define NDIM    4096
#define NGROUPS 32
#define GSIZE   128
#define INV_SCALE 0.0883883476483184406f   // 1/sqrt(128)

// GEMM tiling
#define KC      64                 // K-chunk staged in LDS
#define LDS_STRIDE 72              // halves per row (64 data + 8 pad) -> bank-conflict-free
#define BUF_HALVES (128 * LDS_STRIDE)

typedef _Float16 v16h __attribute__((ext_vector_type(16)));
typedef _Float16 v8h  __attribute__((ext_vector_type(8)));
typedef _Float16 v2h  __attribute__((ext_vector_type(2)));
typedef float    v8f  __attribute__((ext_vector_type(8)));

// ---- WMMA fragment helpers (wave32, 16x16x32 f16) -------------------------
// A (16x32, MxK): lane l holds row M=l&15; elements 0..15 = K[h*16 .. +15],
// h=l>>4
__device__ __forceinline__ v16h frag_a(const _Float16* p) {   // contiguous 16
  v8h lo = *reinterpret_cast<const v8h*>(p);
  v8h hi = *reinterpret_cast<const v8h*>(p + 8);
  return __builtin_shufflevector(lo, hi, 0,1,2,3,4,5,6,7,8,9,10,11,12,13,14,15);
}
// B (32x16, KxN): lane l holds column N=l&15; elements 0..15 = K[h*16 .. +15]
__device__ __forceinline__ v16h frag_b(const _Float16* p) {   // contiguous 16
  v8h lo = *reinterpret_cast<const v8h*>(p);
  v8h hi = *reinterpret_cast<const v8h*>(p + 8);
  return __builtin_shufflevector(lo, hi, 0,1,2,3,4,5,6,7,8,9,10,11,12,13,14,15);
}
// f32-sourced fragments (converted in registers) for the rotation kernel
__device__ __forceinline__ v16h cvt_a_f32(const float* p) {
  v8f lo = *reinterpret_cast<const v8f*>(p);
  v8f hi = *reinterpret_cast<const v8f*>(p + 8);
  v16h r;
#pragma unroll
  for (int i = 0; i < 8; ++i) { r[i] = (_Float16)lo[i]; r[i + 8] = (_Float16)hi[i]; }
  return r;
}
__device__ __forceinline__ v16h cvt_b_f32(const float* p) {
  v8f lo = *reinterpret_cast<const v8f*>(p);
  v8f hi = *reinterpret_cast<const v8f*>(p + 8);
  v16h r;
#pragma unroll
  for (int i = 0; i < 8; ++i) { r[i] = (_Float16)lo[i]; r[i + 8] = (_Float16)hi[i]; }
  return r;
}
__device__ __forceinline__ v8f wmma(v16h a, v16h b, v8f c) {
  return __builtin_amdgcn_wmma_f32_16x16x32_f16(false, a, false, b, (short)0, c,
                                                false, false);
}

// ---- CDNA5 async global->LDS DMA (ASYNCcnt-tracked, bypasses VGPRs) -------
// LDS aperture: LDS address is the low 32 bits of the flat address (ISA 10.2), so the raw LDS
// byte offset is just the truncated flat pointer to a __shared__ object.
__device__ __forceinline__ unsigned lds_addr32(const void* p) {
  return (unsigned)(unsigned long long)p;
}
__device__ __forceinline__ void async_copy_b128(unsigned lds, const _Float16* g) {
  asm volatile("global_load_async_to_lds_b128 %0, %1, off"
               :: "v"(lds), "v"((unsigned long long)g)
               : "memory");
}
__device__ __forceinline__ void wait_async0() {
  asm volatile("s_wait_asynccnt 0x0" ::: "memory");
}

// ---- Kernel 1: dequantize weights (fold norms/scale), f16 out -------------
__global__ __launch_bounds__(256) void dequant_kernel(
    const int* __restrict__ packed, const float* __restrict__ cb,
    const float* __restrict__ norms, _Float16* __restrict__ Wd) {
  const int i = blockIdx.x * 256 + threadIdx.x;      // 0 .. N*K/2-1 (exact grid)
  const int n = i >> 11;                             // K/2 = 2048
  const int j = i & 2047;
  const unsigned byte = (unsigned)packed[i] & 0xFFu;
  const float nrm = norms[(n << 5) + (j >> 6)] * INV_SCALE;  // group = (2j)>>7
  v2h pk;
  pk[0] = (_Float16)(cb[byte & 0xFu] * nrm);         // k = 2j   (low nibble)
  pk[1] = (_Float16)(cb[byte >> 4] * nrm);           // k = 2j+1 (high nibble)
  reinterpret_cast<v2h*>(Wd)[i] = pk;
}

// ---- Kernel 2: per-group rotation via WMMA, f16 out -----------------------
// One wave = 16 tokens x 1 group: C(16x128) = X(16x128) @ R_g^T, 32 WMMAs.
__global__ __launch_bounds__(256) void rotate_kernel(
    const float* __restrict__ x, const float* __restrict__ rot,
    _Float16* __restrict__ Xr) {
  const int lane = threadIdx.x & 31;
  const int wid  = blockIdx.x * 8 + (threadIdx.x >> 5);
  const int g    = wid & (NGROUPS - 1);
  const int t0   = (wid >> 5) * 16;
  const int lq = lane & 15, h = lane >> 4;

  v8f acc[8];
#pragma unroll
  for (int n = 0; n < 8; ++n) acc[n] = {};

  const float* px = x   + (size_t)(t0 + lq) * KDIM + g * GSIZE + h * 16;
  const float* pr = rot + (size_t)g * GSIZE * GSIZE + (size_t)lq * GSIZE + h * 16;

#pragma unroll
  for (int kk = 0; kk < 4; ++kk) {
    const v16h a = cvt_a_f32(px + kk * 32);
#pragma unroll
    for (int n = 0; n < 8; ++n) {
      const v16h b = cvt_b_f32(pr + (size_t)n * 16 * GSIZE + kk * 32);
      acc[n] = wmma(a, b, acc[n]);
    }
  }
#pragma unroll
  for (int n = 0; n < 8; ++n) {
    _Float16* po = Xr + (size_t)(t0 + h * 8) * KDIM + g * GSIZE + n * 16 + lq;
#pragma unroll
    for (int v = 0; v < 8; ++v) po[(size_t)v * KDIM] = (_Float16)acc[n][v];
  }
}

// ---- Kernel 3: main GEMM, async double-buffered through LDS ---------------
// out(8192x4096) = Xr(f16) @ Wd(f16)^T, f32 accum.
// Block: 256 threads = 8 waves (2M x 4N), tile 128x128; wave tile 64x32.
// Per K-chunk (64): stage A(128x64)+B(128x64) f16 via async DMA; 16 WMMAs.
__global__ __launch_bounds__(256) void qgemm_kernel(
    const _Float16* __restrict__ Xr, const _Float16* __restrict__ Wd,
    float* __restrict__ out) {
  __shared__ _Float16 As[2 * BUF_HALVES];
  __shared__ _Float16 Bs[2 * BUF_HALVES];

  const int tid  = threadIdx.x;
  const int lane = tid & 31;
  const int wid  = tid >> 5;
  const int lq = lane & 15, h = lane >> 4;
  const int wm = wid & 1, wn = wid >> 1;
  const int mblk = blockIdx.y * 128;
  const int nblk = blockIdx.x * 128;

  // ---- async copy setup: thread -> (row, 16B segment); 4 passes of 32 rows
  const int rowc = tid >> 3;          // 0..31
  const int seg  = tid & 7;           // 0..7  (seg*8 halves = 16B)
  const _Float16* gA0 = Xr + (size_t)(mblk + rowc) * KDIM + seg * 8;
  const _Float16* gB0 = Wd + (size_t)(nblk + rowc) * KDIM + seg * 8;
  const unsigned laA = lds_addr32(As) + (unsigned)(rowc * LDS_STRIDE + seg * 8) * 2u;
  const unsigned laB = lds_addr32(Bs) + (unsigned)(rowc * LDS_STRIDE + seg * 8) * 2u;
  const unsigned bufBytes  = BUF_HALVES * 2u;
  const unsigned passLds   = 32u * LDS_STRIDE * 2u;

  v8f acc[4][2];
#pragma unroll
  for (int mi = 0; mi < 4; ++mi)
#pragma unroll
    for (int ni = 0; ni < 2; ++ni) acc[mi][ni] = {};

  // prologue: stage chunk 0 into buffer 0
#pragma unroll
  for (int p = 0; p < 4; ++p) {
    async_copy_b128(laA + p * passLds, gA0 + (size_t)p * 32 * KDIM);
    async_copy_b128(laB + p * passLds, gB0 + (size_t)p * 32 * KDIM);
  }

  const int NCHUNK = KDIM / KC;       // 64
  for (int kc = 0; kc < NCHUNK; ++kc) {
    const int buf = kc & 1;
    wait_async0();                    // my copies for chunk kc have landed
    __syncthreads();                  // everyone's copies have landed

    if (kc + 1 < NCHUNK) {            // overlap next chunk's DMA with compute
      const int k1 = (kc + 1) * KC;
      const unsigned lo = ((kc + 1) & 1) * bufBytes;
#pragma unroll
      for (int p = 0; p < 4; ++p) {
        async_copy_b128(laA + lo + p * passLds, gA0 + (size_t)p * 32 * KDIM + k1);
        async_copy_b128(laB + lo + p * passLds, gB0 + (size_t)p * 32 * KDIM + k1);
      }
    }

    const _Float16* a_base = As + buf * BUF_HALVES + (size_t)(wm * 64 + lq) * LDS_STRIDE;
    const _Float16* b_base = Bs + buf * BUF_HALVES + (size_t)(wn * 32 + lq) * LDS_STRIDE;
#pragma unroll
    for (int ks = 0; ks < 2; ++ks) {  // two 32-wide K-steps per chunk
      const int kcol = ks * 32;
      v16h a[4], b[2];
#pragma unroll
      for (int mi = 0; mi < 4; ++mi)
        a[mi] = frag_a(a_base + mi * 16 * LDS_STRIDE + kcol + h * 16);
#pragma unroll
      for (int ni = 0; ni < 2; ++ni)
        b[ni] = frag_b(b_base + ni * 16 * LDS_STRIDE + kcol + h * 16);
#pragma unroll
      for (int mi = 0; mi < 4; ++mi)
#pragma unroll
        for (int ni = 0; ni < 2; ++ni)
          acc[mi][ni] = wmma(a[mi], b[ni], acc[mi][ni]);
    }
    __syncthreads();                  // done reading buf before it is refilled
  }

#pragma unroll
  for (int mi = 0; mi < 4; ++mi)
#pragma unroll
    for (int ni = 0; ni < 2; ++ni) {
      float* po = out + (size_t)(mblk + wm * 64 + mi * 16 + h * 8) * NDIM
                      + nblk + wn * 32 + ni * 16 + lq;
#pragma unroll
      for (int v = 0; v < 8; ++v) po[(size_t)v * NDIM] = acc[mi][ni][v];
    }
}

extern "C" void kernel_launch(void* const* d_in, const int* in_sizes, int n_in,
                              void* d_out, int out_size, void* d_ws, size_t ws_size,
                              hipStream_t stream) {
  (void)in_sizes; (void)n_in; (void)out_size; (void)ws_size;
  const float* x      = (const float*)d_in[0];   // (4,2048,4096) f32
  const int*   packed = (const int*)  d_in[1];   // (4096,2048) int32 (bytes)
  const float* cb     = (const float*)d_in[2];   // (16,) f32
  const float* norms  = (const float*)d_in[3];   // (4096,32) f32
  const float* rot    = (const float*)d_in[4];   // (32,128,128) f32
  float* out = (float*)d_out;                    // (4,2048,4096) f32

  _Float16* Xr = (_Float16*)d_ws;                                       // 64 MiB
  _Float16* Wd = (_Float16*)((char*)d_ws + (size_t)T_TOK * KDIM * 2);   // 32 MiB

  dequant_kernel<<<(NDIM * (KDIM / 2)) / 256, 256, 0, stream>>>(packed, cb, norms, Wd);
  rotate_kernel<<<(T_TOK / 16) * NGROUPS / 8, 256, 0, stream>>>(x, rot, Xr);
  dim3 grid(NDIM / 128, T_TOK / 128);
  qgemm_kernel<<<grid, dim3(256), 0, stream>>>(Xr, Wd, out);
}
